// Leap_batch_3745211482640
// MI455X (gfx1250) — matrix operand order repaired and checked by reading, not compiled, not run on hardware
//
#include <hip/hip_runtime.h>
#include <hip/hip_bf16.h>

// ---------------- problem constants ----------------
#define BB   512
#define SSQ  16
#define ND   40
#define NMED 30
#define EE   64
#define TT   31      // NM+1 decode steps
#define VO   133     // VM+2 output classes
#define VOP  144     // padded to multiple of 16
#define SOS_IDX 131
#define TILE 16      // tokens per block (= WMMA M)

// ---------------- vector types ----------------
typedef _Float16 v16h __attribute__((ext_vector_type(16)));
typedef _Float16 h2   __attribute__((ext_vector_type(2)));
typedef float    v8f  __attribute__((ext_vector_type(8)));
typedef unsigned int v8u __attribute__((ext_vector_type(8)));

static __device__ __forceinline__ unsigned pack2(float a, float b) {
  h2 p; p.x = (_Float16)a; p.y = (_Float16)b;
  return __builtin_bit_cast(unsigned, p);
}
static __device__ __forceinline__ float2 unpack2(unsigned u) {
  h2 p = __builtin_bit_cast(h2, u);
  return make_float2((float)p.x, (float)p.y);
}
static __device__ __forceinline__ float sigmoidf(float x) {
  return 1.0f / (1.0f + __expf(-x));
}

// A fragment: 16x32 f16 (M x K), built from f32 activations in LDS.
// ISA layout: lanes 0-15 M=lane (K base 0), lanes 16-31 M=lane-16 (K base +8);
// VGPR v<4 holds K=2v,2v+1 ; v>=4 holds K=16+2(v-4), +1.
static __device__ __forceinline__ v16h make_afrag(const float* src, int m, int stride,
                                                  int k0, int hi) {
  v8u u;
#pragma unroll
  for (int v = 0; v < 8; ++v) {
    int kv = (v < 4 ? 2 * v : 16 + 2 * (v - 4)) + 8 * hi;
    const float* p = src + m * stride + k0 + kv;
    u[v] = pack2(p[0], p[1]);
  }
  return __builtin_bit_cast(v16h, u);
}

// B fragment: 32x16 f16 (K x N). Weights stored row-major f16 [N][Kdim] in LDS,
// so each VGPR is one contiguous u32 (pair of K values) for column n = lane%16.
// Same K order as A: lanes 0-15 K base 0, lanes 16-31 K base +8; v<4 holds K=2v,2v+1 ; v>=4 holds K=16+2(v-4), +1.
static __device__ __forceinline__ v16h make_bfrag(const _Float16* W, int n, int Kdim,
                                                  int k0, int hi) {
  const unsigned* Wu = (const unsigned*)W;
  v8u u;
#pragma unroll
  for (int v = 0; v < 8; ++v) {
    int kv = (v < 4 ? 2 * v : 16 + 2 * (v - 4)) + 8 * hi;
    u[v] = Wu[(n * Kdim + k0 + kv) >> 1];
  }
  return __builtin_bit_cast(v16h, u);
}

#define WMMA_F16(a, b, c) \
  __builtin_amdgcn_wmma_f32_16x16x32_f16(false, (a), false, (b), (short)0, (c), false, false)

// ---------------- LDS layout (halves first, then floats) ----------------
#define WIH_H (192 * 128)
#define WHH_H (192 * 64)
#define WO_HN (VOP * 64)
#define EMB_H (TILE * ND * EE)
#define LDS_FLOATS (192 + 192 + VOP + EE + EE + TILE*ND + TILE*EE + TILE*2*EE + \
                    TILE*ND + TILE*8 + TILE + 2*TILE + TILE*VOP)
#define SHMEM_BYTES ((size_t)((WIH_H + WHH_H + WO_HN + EMB_H) * 2 + LDS_FLOATS * 4))

__global__ __launch_bounds__(128) void leap_gru_kernel(
    const int* __restrict__ diseases, const int* __restrict__ medications,
    const float* __restrict__ d_mask, const float* __restrict__ diag_table,
    const float* __restrict__ med_table, const float* __restrict__ attn_w,
    const float* __restrict__ attn_b, const float* __restrict__ w_ih,
    const float* __restrict__ w_hh, const float* __restrict__ b_ih,
    const float* __restrict__ b_hh, const float* __restrict__ wo_w,
    const float* __restrict__ wo_b, float* __restrict__ out) {
  extern __shared__ __align__(16) char smem[];
  _Float16* sWih = (_Float16*)smem;           // [192][128]
  _Float16* sWhh = sWih + WIH_H;              // [192][64]
  _Float16* sWo  = sWhh + WHH_H;              // [144][64] (pad rows zero)
  _Float16* sEmb = sWo + WO_HN;               // [16][40][64]
  float* sBih  = (float*)(sEmb + EMB_H);      // 192
  float* sBhh  = sBih + 192;                  // 192
  float* sBo   = sBhh + 192;                  // 144
  float* sWh   = sBo + VOP;                   // 64
  float* sWd   = sWh + EE;                    // 64
  float* sDb   = sWd + EE;                    // 16*40 score base
  float* sH    = sDb + TILE * ND;             // 16*64 hidden state f32
  float* sX    = sH + TILE * EE;              // 16*128 [ctx | dec]
  float* sP    = sX + TILE * 2 * EE;          // 16*40 attn (unnormalized exp)
  float* sRed  = sP + TILE * ND;              // 16*8 partials
  float* sInv  = sRed + TILE * 8;             // 16 softmax 1/sum
  float* sStat = sInv + TILE;                 // 16*2 max / lse
  float* sLg   = sStat + 2 * TILE;            // 16*144 logits

  const int tid = threadIdx.x;
  const int tile0 = blockIdx.x * TILE;

  // ---------------- setup: stage weights (f32 -> f16) ----------------
  for (int i = tid; i < WIH_H / 2; i += 128)
    ((unsigned*)sWih)[i] = pack2(w_ih[2 * i], w_ih[2 * i + 1]);
  for (int i = tid; i < WHH_H / 2; i += 128)
    ((unsigned*)sWhh)[i] = pack2(w_hh[2 * i], w_hh[2 * i + 1]);
  for (int i = tid; i < WO_HN / 2; i += 128) {
    int n = (2 * i) >> 6, k = (2 * i) & 63;
    float a = (n < VO) ? wo_w[n * EE + k] : 0.0f;
    float b = (n < VO) ? wo_w[n * EE + k + 1] : 0.0f;
    ((unsigned*)sWo)[i] = pack2(a, b);
  }
  for (int i = tid; i < 192; i += 128) { sBih[i] = b_ih[i]; sBhh[i] = b_hh[i]; }
  for (int i = tid; i < VOP; i += 128) sBo[i] = (i < VO) ? wo_b[i] : 0.0f;
  for (int i = tid; i < EE; i += 128) { sWh[i] = attn_w[i]; sWd[i] = attn_w[EE + i]; }
  // h0 = med_table[SOS] broadcast to all tokens
  for (int i = tid; i < TILE * EE; i += 128) sH[i] = med_table[SOS_IDX * EE + (i & 63)];
  // gather disease embeddings -> f16 LDS
  for (int i = tid; i < EMB_H / 2; i += 128) {
    int row = i >> 5, c = i & 31;
    int m = row / ND, n = row % ND;
    int didx = diseases[(tile0 + m) * ND + n];
    const float* src = diag_table + (size_t)didx * EE + 2 * c;
    ((unsigned*)sEmb)[i] = pack2(src[0], src[1]);
  }
  __syncthreads();
  // d_base = emb . w_d + mask + attn_b
  for (int r = tid; r < TILE * ND; r += 128) {
    int m = r / ND, n = r % ND;
    const unsigned* eh = (const unsigned*)sEmb + r * 32;
    float acc = attn_b[0] + d_mask[(tile0 + m) * ND + n];
#pragma unroll
    for (int c = 0; c < 32; ++c) {
      float2 v = unpack2(eh[c]);
      acc += v.x * sWd[2 * c] + v.y * sWd[2 * c + 1];
    }
    sDb[r] = acc;
  }
  __syncthreads();

  // per-wave constants (wave-uniform branches keep EXEC all-ones for WMMA)
  const int lane = tid & 31, wv = tid >> 5, nl = lane & 15, hi = lane >> 4;
  const int nR = wv * 16, nZ = (wv + 4) * 16, nN = (wv + 8) * 16;
  const int eBase = wv * 16 + nl;  // hidden index owned by this lane's D column

  for (int t = 0; t < TT; ++t) {
    // -------- 1a: hdot partials --------
    {
      int m = tid & 15, c = tid >> 4;
      float acc = 0.f;
#pragma unroll
      for (int j = 0; j < 8; ++j) acc += sH[m * EE + c * 8 + j] * sWh[c * 8 + j];
      sRed[m * 8 + c] = acc;
    }
    __syncthreads();
    // -------- 1b: softmax (unnormalized; 1/sum folded into ctx) --------
    if (tid < TILE) {
      int m = tid;
      float hd = 0.f;
#pragma unroll
      for (int c = 0; c < 8; ++c) hd += sRed[m * 8 + c];
      float mx = -3.0e38f;
      for (int n = 0; n < ND; ++n) mx = fmaxf(mx, sDb[m * ND + n] + hd);
      float sum = 0.f;
      for (int n = 0; n < ND; ++n) {
        float e = __expf(sDb[m * ND + n] + hd - mx);
        sP[m * ND + n] = e;
        sum += e;
      }
      sInv[m] = 1.0f / sum;
    }
    __syncthreads();
    // -------- 2: ctx = attn @ emb, plus decoder-input gather --------
    {
      int m = tid >> 3, g = tid & 7;
      float acc[8];
#pragma unroll
      for (int j = 0; j < 8; ++j) acc[j] = 0.f;
      const unsigned* eh = (const unsigned*)sEmb + m * ND * 32 + g * 4;
      for (int n = 0; n < ND; ++n) {
        float p = sP[m * ND + n];
        const unsigned* e2 = eh + n * 32;
#pragma unroll
        for (int c = 0; c < 4; ++c) {
          float2 v = unpack2(e2[c]);
          acc[2 * c] += p * v.x;
          acc[2 * c + 1] += p * v.y;
        }
      }
      float inv = sInv[m];
#pragma unroll
      for (int j = 0; j < 8; ++j) sX[m * 2 * EE + g * 8 + j] = acc[j] * inv;
    }
    for (int i = tid; i < TILE * EE; i += 128) {
      int m = i >> 6, e = i & 63;
      int row = (t == 0) ? SOS_IDX : medications[(tile0 + m) * NMED + (t - 1)];
      sX[m * 2 * EE + EE + e] = med_table[row * EE + e];
    }
    __syncthreads();
    // -------- 3: WMMA — gi = x@W_ih^T + b_ih ; gh = h@W_hh^T + b_hh --------
    v8f cR, cZ, cN, dR, dZ, dN;
    {
      float v0 = sBih[nR + nl], v1 = sBih[nZ + nl], v2 = sBih[nN + nl];
      float v3 = sBhh[nR + nl], v4 = sBhh[nZ + nl], v5 = sBhh[nN + nl];
#pragma unroll
      for (int r = 0; r < 8; ++r) {
        cR[r] = v0; cZ[r] = v1; cN[r] = v2;
        dR[r] = v3; dZ[r] = v4; dN[r] = v5;
      }
    }
#pragma unroll
    for (int kt = 0; kt < 4; ++kt) {
      v16h a = make_afrag(sX, nl, 2 * EE, 32 * kt, hi);
      cR = WMMA_F16(a, make_bfrag(sWih, nR + nl, 2 * EE, 32 * kt, hi), cR);
      cZ = WMMA_F16(a, make_bfrag(sWih, nZ + nl, 2 * EE, 32 * kt, hi), cZ);
      cN = WMMA_F16(a, make_bfrag(sWih, nN + nl, 2 * EE, 32 * kt, hi), cN);
    }
#pragma unroll
    for (int kt = 0; kt < 2; ++kt) {
      v16h a = make_afrag(sH, nl, EE, 32 * kt, hi);
      dR = WMMA_F16(a, make_bfrag(sWhh, nR + nl, EE, 32 * kt, hi), dR);
      dZ = WMMA_F16(a, make_bfrag(sWhh, nZ + nl, EE, 32 * kt, hi), dZ);
      dN = WMMA_F16(a, make_bfrag(sWhh, nN + nl, EE, 32 * kt, hi), dN);
    }
    __syncthreads();  // all reads of sH / sX complete before h update
    // -------- 4: GRU elementwise update (D-layouts line up by construction) --------
#pragma unroll
    for (int r = 0; r < 8; ++r) {
      int m = r + 8 * hi;
      float rg = sigmoidf(cR[r] + dR[r]);
      float zg = sigmoidf(cZ[r] + dZ[r]);
      float ng = tanhf(cN[r] + rg * dN[r]);
      float ho = sH[m * EE + eBase];
      sH[m * EE + eBase] = (1.0f - zg) * ng + zg * ho;
    }
    __syncthreads();
    // -------- 5: logits = h_new @ wo_w^T + wo_b (N padded to 144) --------
    {
      v16h a0 = make_afrag(sH, nl, EE, 0, hi);
      v16h a1 = make_afrag(sH, nl, EE, 32, hi);
      auto logit_tile = [&](int nt) {
        int n0 = nt * 16;
        float bo = sBo[n0 + nl];
        v8f c;
#pragma unroll
        for (int r = 0; r < 8; ++r) c[r] = bo;
        c = WMMA_F16(a0, make_bfrag(sWo, n0 + nl, EE, 0, hi), c);
        c = WMMA_F16(a1, make_bfrag(sWo, n0 + nl, EE, 32, hi), c);
#pragma unroll
        for (int r = 0; r < 8; ++r) sLg[(r + 8 * hi) * VOP + n0 + nl] = c[r];
      };
      logit_tile(wv);
      logit_tile(wv + 4);
      if (wv == 0) logit_tile(8);  // wave-uniform: EXEC stays all-ones
    }
    __syncthreads();
    // -------- 6: log-softmax stats over the 133 real classes --------
    if (tid < TILE) {
      int m = tid;
      float mx = -3.0e38f;
      for (int n = 0; n < VO; ++n) mx = fmaxf(mx, sLg[m * VOP + n]);
      float sum = 0.f;
      for (int n = 0; n < VO; ++n) sum += __expf(sLg[m * VOP + n] - mx);
      sStat[2 * m] = mx;
      sStat[2 * m + 1] = __logf(sum);
    }
    __syncthreads();
    // -------- 7: write out[(token*31 + t)*133 + n] --------
    for (int i = tid; i < TILE * VO; i += 128) {
      int m = i / VO, n = i - m * VO;
      float v = sLg[m * VOP + n] - sStat[2 * m] - sStat[2 * m + 1];
      out[((size_t)(tile0 + m) * TT + t) * VO + n] = v;
    }
    __syncthreads();
  }
}

extern "C" void kernel_launch(void* const* d_in, const int* in_sizes, int n_in,
                              void* d_out, int out_size, void* d_ws, size_t ws_size,
                              hipStream_t stream) {
  (void)in_sizes; (void)n_in; (void)out_size; (void)d_ws; (void)ws_size;
  const int*   diseases    = (const int*)d_in[0];
  // d_in[1] = procedures (unused by reference)
  const int*   medications = (const int*)d_in[2];
  const float* d_mask      = (const float*)d_in[3];
  // d_in[4] = m_mask_matrix (unused), d_in[5] = seq_length (unused)
  const float* diag_table  = (const float*)d_in[6];
  const float* med_table   = (const float*)d_in[7];
  const float* attn_w      = (const float*)d_in[8];
  const float* attn_b      = (const float*)d_in[9];
  const float* w_ih        = (const float*)d_in[10];
  const float* w_hh        = (const float*)d_in[11];
  const float* b_ih        = (const float*)d_in[12];
  const float* b_hh        = (const float*)d_in[13];
  const float* wo_w        = (const float*)d_in[14];
  const float* wo_b        = (const float*)d_in[15];
  float* out = (float*)d_out;

  (void)hipFuncSetAttribute(reinterpret_cast<const void*>(leap_gru_kernel),
                            hipFuncAttributeMaxDynamicSharedMemorySize,
                            (int)SHMEM_BYTES);
  dim3 grid((BB * SSQ) / TILE);  // 512 tiles of 16 tokens
  dim3 block(128);               // 4 wave32 per tile
  leap_gru_kernel<<<grid, block, SHMEM_BYTES, stream>>>(
      diseases, medications, d_mask, diag_table, med_table, attn_w, attn_b,
      w_ih, w_hh, b_ih, b_hh, wo_w, wo_b, out);
}
